// MHA_56667798504120
// MI455X (gfx1250) — matrix operand order repaired and checked by reading, not compiled, not run on hardware
//
#include <hip/hip_runtime.h>
#include <hip/hip_bf16.h>

// MHA forward for MI455X (gfx1250, wave32).
// All GEMMs via v_wmma_f32_16x16x32_bf16 (bf16 in, f32 accumulate).
// - Projection GEMMs: 2x2 register-blocked wave tiles (32x32 per wave):
//   8 fragment loads -> 4 WMMAs (2x the arithmetic intensity of 1x1 tiles).
// - Attention fused per 16-row q-tile: scores (Q frag hoisted, K prefetch)
//   -> softmax in LDS -> single attn write -> split-K AV across all 8 waves.
// attn [B,H,S,S] (537 MB) is written exactly once and never re-read.

typedef __bf16 bf16_t;
typedef __attribute__((ext_vector_type(16))) __bf16 v16bf;
typedef __attribute__((ext_vector_type(8)))  __bf16 v8bf;
typedef __attribute__((ext_vector_type(8)))  float  v8f;

#define B_   2
#define S_   2048
#define D_   1024
#define H_   16
#define DH_  64
#define NROW (B_ * S_)   // 4096 token rows

// ---- WMMA fragment loaders (layouts per CDNA5 ISA §7.12.2) ----
// A 16x32 bf16: lane l -> row m=l&15; half=l>>4; halfs K = 8*half+{0..7}, then +16.
__device__ __forceinline__ v16bf load_fragA(const bf16_t* p) {
  v8bf lo = *(const v8bf*)(p);
  v8bf hi = *(const v8bf*)(p + 16);
  v16bf r;
#pragma unroll
  for (int i = 0; i < 8; ++i) { r[i] = lo[i]; r[i + 8] = hi[i]; }
  return r;
}
// B 32x16 bf16 (loaded from B^T row-major): lane l -> col n=l&15; halfs
// K = 8*half+{0..7}, then +16.
__device__ __forceinline__ v16bf load_fragB(const bf16_t* p) {
  v8bf lo = *(const v8bf*)(p);
  v8bf hi = *(const v8bf*)(p + 16);
  v16bf r;
#pragma unroll
  for (int i = 0; i < 8; ++i) { r[i] = lo[i]; r[i + 8] = hi[i]; }
  return r;
}

__device__ __forceinline__ v8f wmma_bf16(v16bf a, v16bf b, v8f c) {
  return __builtin_amdgcn_wmma_f32_16x16x32_bf16(false, a, false, b,
                                                 (short)0, c, false, false);
}

// One wave, one 16x16 tile, K range [k0,k1). C/D: c[r] -> m=r+8*(lane>=16), n=lane&15.
__device__ __forceinline__ v8f wmma_range(const bf16_t* A, int lda,
                                          const bf16_t* Bt, int ldb,
                                          int k0, int k1) {
  const int lane = threadIdx.x & 31;
  const int r  = lane & 15;
  const int hl = lane >> 4;
  const bf16_t* ap = A  + (size_t)r * lda + 8  * hl;
  const bf16_t* bp = Bt + (size_t)r * ldb + 8  * hl;
  v8f c = {};
  for (int k = k0; k < k1; k += 32)
    c = wmma_bf16(load_fragA(ap + k), load_fragB(bp + k), c);
  return c;
}

// One wave, 32x32 macro-tile (2x2 register blocking): 8 loads feed 4 WMMAs/step.
__device__ __forceinline__ void gemm2x2(const bf16_t* A, int lda,
                                        const bf16_t* Bt, int ldb, int Kdim,
                                        v8f cc[2][2]) {
  const int lane = threadIdx.x & 31;
  const int r  = lane & 15;
  const int hl = lane >> 4;
  const bf16_t* ap0 = A  + (size_t)r * lda + 8 * hl;
  const bf16_t* ap1 = ap0 + (size_t)16 * lda;
  const bf16_t* bp0 = Bt + (size_t)r * ldb + 8 * hl;
  const bf16_t* bp1 = bp0 + (size_t)16 * ldb;
  for (int k = 0; k < Kdim; k += 32) {
    v16bf a0 = load_fragA(ap0 + k);
    v16bf a1 = load_fragA(ap1 + k);
    v16bf b0 = load_fragB(bp0 + k);
    v16bf b1 = load_fragB(bp1 + k);
    cc[0][0] = wmma_bf16(a0, b0, cc[0][0]);
    cc[0][1] = wmma_bf16(a0, b1, cc[0][1]);
    cc[1][0] = wmma_bf16(a1, b0, cc[1][0]);
    cc[1][1] = wmma_bf16(a1, b1, cc[1][1]);
  }
}

// ---- f32 -> bf16 conversion pass ----
__global__ void mha_cvt_bf16_kernel(const float* __restrict__ src,
                                    bf16_t* __restrict__ dst, int n) {
  int i = blockIdx.x * blockDim.x + threadIdx.x;
  const int stride = gridDim.x * blockDim.x;
  for (; i < n; i += stride) dst[i] = (bf16_t)src[i];
}

// ---- fused QKV projection: Y = X @ W^T + b, blockIdx.z selects Q/K/V ----
// Q head-major [b][h][s][dh], pre-scaled by 1/sqrt(dh)=0.125.
// K head-major [b][h][s][dh]. V transposed [b][h][dh][s].
__global__ void mha_qkv_proj_kernel(const bf16_t* __restrict__ Xq, const bf16_t* __restrict__ Xk,
                                    const bf16_t* __restrict__ Xv,
                                    const bf16_t* __restrict__ Wq, const bf16_t* __restrict__ Wk,
                                    const bf16_t* __restrict__ Wv,
                                    const float* __restrict__ bq, const float* __restrict__ bk,
                                    const float* __restrict__ bv,
                                    bf16_t* __restrict__ Qh, bf16_t* __restrict__ Kh,
                                    bf16_t* __restrict__ Vt) {
  const int which = blockIdx.z;
  const bf16_t* X   = which == 0 ? Xq : (which == 1 ? Xk : Xv);
  const bf16_t* W   = which == 0 ? Wq : (which == 1 ? Wk : Wv);
  const float* bias = which == 0 ? bq : (which == 1 ? bk : bv);
  const int wave = threadIdx.x >> 5;
  const int lane = threadIdx.x & 31;
  const int row0 = (blockIdx.y * 2 + (wave >> 2)) * 32;   // token macro-tile
  const int col0 = (blockIdx.x * 4 + (wave & 3)) * 32;    // feature macro-tile
  v8f cc[2][2] = {};
  gemm2x2(X + (size_t)row0 * D_, D_, W + (size_t)col0 * D_, D_, D_, cc);
  const int n  = lane & 15;
  const int mh = (lane >> 4) * 8;
#pragma unroll
  for (int i = 0; i < 2; ++i) {
#pragma unroll
    for (int j = 0; j < 2; ++j) {
      const int col = col0 + 16 * j + n;
      const int h  = col >> 6;
      const int dh = col & 63;
      const float bb = bias[col];
#pragma unroll
      for (int r = 0; r < 8; ++r) {
        const int m = row0 + 16 * i + mh + r;
        const int b = m >> 11;
        const int s = m & 2047;
        const float y = cc[i][j][r] + bb;
        if (which == 0)
          Qh[(((size_t)b * H_ + h) * S_ + s) * DH_ + dh] = (bf16_t)(y * 0.125f);
        else if (which == 1)
          Kh[(((size_t)b * H_ + h) * S_ + s) * DH_ + dh] = (bf16_t)y;
        else
          Vt[(((size_t)b * H_ + h) * DH_ + dh) * (size_t)S_ + s] = (bf16_t)y;
      }
    }
  }
}

// ---- fused scores -> softmax -> attn write -> split-K AV, per (qtile,h,b) ----
// Block = 256 threads = 8 waves. Dyn LDS = 16*2048*4 (f32) + 16*2048*2 (bf16).
__global__ void mha_attn_kernel(const bf16_t* __restrict__ Qh, const bf16_t* __restrict__ Kh,
                                const bf16_t* __restrict__ Vt, float* __restrict__ attn,
                                bf16_t* __restrict__ ctx) {
  extern __shared__ char smem[];
  float*  sc = (float*)smem;                                    // [16][2048] f32
  bf16_t* pb = (bf16_t*)(smem + 16 * 2048 * sizeof(float));     // [16][2048] bf16
  const int qt = blockIdx.x, h = blockIdx.y, b = blockIdx.z;
  const int q0 = qt * 16;
  const int ntile = qt + 1;          // causal: only key-tiles 0..qt are live
  const int wave = threadIdx.x >> 5;
  const int lane = threadIdx.x & 31;
  const int rr = lane & 15;
  const int hl = lane >> 4;
  const bf16_t* Q = Qh + (((size_t)b * H_ + h) * S_ + q0) * DH_;
  const bf16_t* K = Kh + ((size_t)b * H_ + h) * S_ * DH_;

  // scores: S_tile = (Q/8) @ K_tile^T.  Q fragment (2 k-steps) hoisted: it is
  // invariant across all key tiles, so each tile costs 2 B loads for 2 WMMAs.
  const bf16_t* qp = Q + (size_t)rr * DH_ + 8 * hl;
  const v16bf qa0 = load_fragA(qp);
  const v16bf qa1 = load_fragA(qp + 32);
  for (int t = wave; t < ntile; t += 8) {
    const bf16_t* kp = K + (size_t)t * 16 * DH_ + (size_t)rr * DH_ + 8 * hl;
    if (t + 8 < ntile) __builtin_prefetch(kp + 8 * 16 * DH_, 0, 1);  // global_prefetch_b8
    v8f c = {};
    c = wmma_bf16(qa0, load_fragB(kp), c);
    c = wmma_bf16(qa1, load_fragB(kp + 32), c);
    const int n = lane & 15, mh = hl * 8;
#pragma unroll
    for (int r = 0; r < 8; ++r) sc[(mh + r) * 2048 + t * 16 + n] = c[r];
  }
  __syncthreads();

  // softmax: 16 threads own each of the 16 rows; row i valid over [0, q0+i].
  const int row = threadIdx.x >> 4;
  const int tl  = threadIdx.x & 15;
  const int len = q0 + row + 1;
  float* srow = sc + row * 2048;
  float mx = -3.0e38f;
  for (int cI = tl; cI < len; cI += 16) mx = fmaxf(mx, srow[cI]);
#pragma unroll
  for (int off = 8; off; off >>= 1) mx = fmaxf(mx, __shfl_xor(mx, off, 16));
  float sum = 0.f;
  for (int cI = tl; cI < len; cI += 16) {
    const float e = __expf(srow[cI] - mx);
    srow[cI] = e;
    sum += e;
  }
#pragma unroll
  for (int off = 8; off; off >>= 1) sum += __shfl_xor(sum, off, 16);
  const float inv = 1.f / sum;

  // single write of attn probs + bf16 pack for the AV wmma
  float*  orow = attn + (((size_t)b * H_ + h) * S_ + (q0 + row)) * (size_t)S_;
  bf16_t* prow = pb + row * 2048;
  for (int cI = tl; cI < S_; cI += 16) {
    const float p = (cI < len) ? srow[cI] * inv : 0.f;
    orow[cI] = p;
    prow[cI] = (bf16_t)p;
  }
  __syncthreads();

  // AV: ctx(16x64) = P(16xKred) @ V(Kredx64).  All 8 waves: 4 n-tiles x 2 K
  // halves; the upper-half partials go through LDS (score region is dead now).
  const int ntl = wave & 3;          // n-tile 0..3
  const int kh  = wave >> 2;         // K half
  const int n0  = ntl * 16;
  const int Kred   = (ntile * 16 + 31) & ~31;  // probs are exact zeros beyond len
  const int nk     = Kred >> 5;
  const int ksplit = ((nk + 1) >> 1) * 32;
  const int k0 = kh ? ksplit : 0;
  const int k1 = kh ? Kred : ksplit;
  const bf16_t* Vth = Vt + (((size_t)b * H_ + h) * DH_ + n0) * (size_t)S_;
  v8f c = wmma_range(pb, 2048, Vth, 2048, k0, k1);
  const int n = lane & 15, mh = hl * 8;
  float* part = sc;                  // reuse: [4 n-tiles][16][16] f32 partials
  if (kh) {
#pragma unroll
    for (int r = 0; r < 8; ++r) part[ntl * 256 + (mh + r) * 16 + n] = c[r];
  }
  __syncthreads();
  if (!kh) {
#pragma unroll
    for (int r = 0; r < 8; ++r) {
      const int s = q0 + mh + r;
      const float y = c[r] + part[ntl * 256 + (mh + r) * 16 + n];
      ctx[((size_t)b * S_ + s) * D_ + h * DH_ + n0 + n] = (bf16_t)y;
    }
  }
}

// ---- output projection: y = ctx @ Wo^T + bo (f32 out), 2x2 blocked ----
__global__ void mha_out_proj_kernel(const bf16_t* __restrict__ Ctx, const bf16_t* __restrict__ Wo,
                                    const float* __restrict__ bo, float* __restrict__ Y) {
  const int wave = threadIdx.x >> 5;
  const int lane = threadIdx.x & 31;
  const int row0 = (blockIdx.y * 2 + (wave >> 2)) * 32;
  const int col0 = (blockIdx.x * 4 + (wave & 3)) * 32;
  v8f cc[2][2] = {};
  gemm2x2(Ctx + (size_t)row0 * D_, D_, Wo + (size_t)col0 * D_, D_, D_, cc);
  const int n  = lane & 15;
  const int mh = (lane >> 4) * 8;
#pragma unroll
  for (int i = 0; i < 2; ++i) {
#pragma unroll
    for (int j = 0; j < 2; ++j) {
      const float bb = bo[col0 + 16 * j + n];
#pragma unroll
      for (int r = 0; r < 8; ++r)
        Y[(size_t)(row0 + 16 * i + mh + r) * D_ + col0 + 16 * j + n] = cc[i][j][r] + bb;
    }
  }
}

extern "C" void kernel_launch(void* const* d_in, const int* in_sizes, int n_in,
                              void* d_out, int out_size, void* d_ws, size_t ws_size,
                              hipStream_t stream) {
  (void)in_sizes; (void)n_in; (void)out_size; (void)ws_size;
  const float* q  = (const float*)d_in[0];
  const float* k  = (const float*)d_in[1];
  const float* v  = (const float*)d_in[2];
  // d_in[3] = causal mask (implemented analytically)
  const float* wq = (const float*)d_in[4];
  const float* bq = (const float*)d_in[5];
  const float* wk = (const float*)d_in[6];
  const float* bk = (const float*)d_in[7];
  const float* wv = (const float*)d_in[8];
  const float* bv = (const float*)d_in[9];
  const float* wo = (const float*)d_in[10];
  const float* bo = (const float*)d_in[11];

  const size_t NE = (size_t)NROW * D_;   // 4,194,304 activation elems
  const size_t WE = (size_t)D_ * D_;     // 1,048,576 weight elems
  bf16_t* ws  = (bf16_t*)d_ws;           // total 7*NE + 4*WE bf16 = 64 MB
  bf16_t* xq  = ws;
  bf16_t* xk  = xq + NE;
  bf16_t* xv  = xk + NE;
  bf16_t* wqb = xv + NE;
  bf16_t* wkb = wqb + WE;
  bf16_t* wvb = wkb + WE;
  bf16_t* wob = wvb + WE;
  bf16_t* Qh  = wob + WE;
  bf16_t* Kh  = Qh + NE;
  bf16_t* Vt  = Kh + NE;
  bf16_t* ctx = Vt + NE;

  float* y    = (float*)d_out;           // [B,S,D]
  float* attn = y + NE;                  // [B,H,S,S]

  mha_cvt_bf16_kernel<<<2048, 256, 0, stream>>>(q,  xq,  (int)NE);
  mha_cvt_bf16_kernel<<<2048, 256, 0, stream>>>(k,  xk,  (int)NE);
  mha_cvt_bf16_kernel<<<2048, 256, 0, stream>>>(v,  xv,  (int)NE);
  mha_cvt_bf16_kernel<<<1024, 256, 0, stream>>>(wq, wqb, (int)WE);
  mha_cvt_bf16_kernel<<<1024, 256, 0, stream>>>(wk, wkb, (int)WE);
  mha_cvt_bf16_kernel<<<1024, 256, 0, stream>>>(wv, wvb, (int)WE);
  mha_cvt_bf16_kernel<<<1024, 256, 0, stream>>>(wo, wob, (int)WE);

  // block tile 64 rows x 128 cols (8 waves of 32x32); z = Q/K/V
  mha_qkv_proj_kernel<<<dim3(8, 64, 3), 256, 0, stream>>>(
      xq, xk, xv, wqb, wkb, wvb, bq, bk, bv, Qh, Kh, Vt);

  // one block per (16-row q-tile, head, batch); 192 KB dynamic LDS
  mha_attn_kernel<<<dim3(S_ / 16, H_, B_), 256,
                    16 * 2048 * (sizeof(float) + sizeof(bf16_t)), stream>>>(
      Qh, Kh, Vt, attn, ctx);

  mha_out_proj_kernel<<<dim3(8, 64, 1), 256, 0, stream>>>(ctx, wob, bo, y);
}
